// Mamba2_37228776522457
// MI455X (gfx1250) — matrix operand order repaired and checked by reading, not compiled, not run on hardware
//
#include <hip/hip_runtime.h>
#include <hip/hip_bf16.h>

// ---------------------------------------------------------------------------
// Mamba2 forward for MI455X (gfx1250, wave32, WMMA + Tensor Data Mover).
// Heavy matmuls: v_wmma_f32_16x16x32_bf16, f32 accumulate.
// Tile staging:  tensor_load_to_lds (TDM) double-buffered, s_wait_tensorcnt.
// ---------------------------------------------------------------------------

typedef __bf16 bf16;
typedef __attribute__((ext_vector_type(8)))  __bf16 bf16x8;
typedef __attribute__((ext_vector_type(16))) __bf16 bf16x16;
typedef __attribute__((ext_vector_type(8)))  float  f32x8;
typedef __attribute__((ext_vector_type(4)))  unsigned int u32x4;
typedef __attribute__((ext_vector_type(8)))  int   i32x8;
typedef __attribute__((ext_vector_type(4)))  int   i32x4;

#define HIDDEN   2048
#define INTER    4096
#define NHEADS   64
#define HEADDIM  64
#define STATE    128
#define CONVK    4
#define CHUNK    256
#define NCHUNK   16
#define CONV_DIM 4352            // INTER + 2*STATE
#define PROJ_N   8512            // INTER + CONV_DIM + NHEADS
#define SEQ      4096

#if __has_builtin(__builtin_amdgcn_tensor_load_to_lds)
#define HAVE_TDM 1
#else
#define HAVE_TDM 0
#endif

// ---- WMMA fragment loaders (wave32 layouts, CDNA5 ISA 7.12.2) --------------
// A-matrix 16x32 bf16: lane L: M = L&15, half h = L>>4; one contiguous
// 16-element K-run at k0 + 16h.
__device__ inline bf16x16 ldsA(const bf16* p /* &lds[row*ldk + k0] */, int hl) {
    const bf16* q = p + 16 * hl;
    bf16x8 lo = *(const bf16x8*)q;
    bf16x8 hi = *(const bf16x8*)(q + 8);
    bf16x16 o;
#pragma unroll
    for (int i = 0; i < 8; ++i) { o[i] = lo[i]; o[8 + i] = hi[i]; }
    return o;
}
// B-matrix 32x16 bf16 stored [N][K]: lane L: N = L&15, h = L>>4; one
// contiguous 16-element K-run at k0 + 16h.
__device__ inline bf16x16 ldsB(const bf16* p /* &lds[n*ldk + k0] */, int hl) {
    const bf16* q = p + 16 * hl;
    bf16x8 lo = *(const bf16x8*)q;
    bf16x8 hi = *(const bf16x8*)(q + 8);
    bf16x16 o;
#pragma unroll
    for (int i = 0; i < 8; ++i) { o[i] = lo[i]; o[8 + i] = hi[i]; }
    return o;
}

__device__ inline f32x8 wmma_bf16(bf16x16 a, bf16x16 b, f32x8 c) {
    return __builtin_amdgcn_wmma_f32_16x16x32_bf16(false, a, false, b,
                                                   (short)0, c, false, false);
}

__device__ inline float silu(float x) { return x / (1.f + __expf(-x)); }

// Workgroup-relative LDS byte address (flat shared addr low 32 bits).
__device__ inline unsigned lds_addr32(const void* p) {
    return (unsigned)(unsigned long long)(__SIZE_TYPE__)p;
}

// ---- Tensor Data Mover: 2D tile global -> LDS ------------------------------
// rows x row_bytes tile, global row stride stride_bytes, data_size = 1 byte.
// padflags: bits for D# group1 word0 (pad_enable/pad_interval/pad_amount).
__device__ inline void tdm_load_2d(void* lds_ptr, const void* gptr,
                                   unsigned row_bytes, unsigned rows,
                                   unsigned stride_bytes, unsigned padflags)
{
#if HAVE_TDM
    unsigned long long ga = (unsigned long long)gptr;
    u32x4 g0 = { 1u,                                   // count = 1 valid D#
                 lds_addr32(lds_ptr),                  // lds_addr
                 (unsigned)ga,                         // global_addr[31:0]
                 ((unsigned)(ga >> 32) & 0x1FFFFFFu)   // global_addr[56:32]
                   | (2u << 30) };                     // type = 2 (image)
    i32x8 g1 = { (int)padflags,                        // data_size=0 (1B), pads
                 (int)(row_bytes << 16),               // tensor_dim0[15:0]
                 (int)(rows << 16),                    // tensor_dim1[15:0]
                 (int)(row_bytes << 16),               // tile_dim0
                 (int)rows,                            // tile_dim1
                 (int)stride_bytes,                    // tensor_dim0_stride
                 0, 0 };
    i32x4 z4 = { 0, 0, 0, 0 };
    i32x8 z8 = { 0, 0, 0, 0, 0, 0, 0, 0 };
    __builtin_amdgcn_tensor_load_to_lds(g0, g1, z4, z4, z8, 0);
#else
    // fallback: calling-wave cooperative copy (slow, compile-safe)
    const char* src = (const char*)gptr;
    char* dst = (char*)lds_ptr;
    const unsigned lrow = row_bytes + (((padflags >> 20) & 1u) ? 16u : 0u);
    const int lane = threadIdx.x & 31;
    for (unsigned row = 0; row < rows; ++row)
        for (unsigned b = (unsigned)lane * 4u; b < row_bytes; b += 128u)
            *(unsigned*)(dst + (size_t)row * lrow + b) =
                *(const unsigned*)(src + (size_t)row * stride_bytes + b);
#endif
}

// ---------------------------------------------------------------------------
// fp32 -> bf16 convert
// ---------------------------------------------------------------------------
__global__ void k_to_bf16(const float* __restrict__ in, bf16* __restrict__ out, int n) {
    int i = blockIdx.x * 256 + threadIdx.x;
    if (i < n) out[i] = (bf16)in[i];
}

// ---------------------------------------------------------------------------
// GEMM: C[M][N] = A[M][K] @ B[N][K]^T   (A,B bf16 row-major, C fp32)
// 256 thr (8 waves), tile 128x128, K-step 32, TDM double-buffered staging.
// ---------------------------------------------------------------------------
#define LDK 40   // LDS K-stride: 32 data + 8 pad (TDM pads 16B per 64B row)
#define GEMM_PAD ((1u << 20) | (3u << 22) | (3u << 25)) // pad 4 dw / 16 dw

__global__ __launch_bounds__(256) void k_gemm_bf16(
    const bf16* __restrict__ A, const bf16* __restrict__ B,
    float* __restrict__ C, int M, int N, int K)
{
    __shared__ __align__(16) bf16 sAb[2][128 * LDK];
    __shared__ __align__(16) bf16 sBb[2][128 * LDK];

    const int tid = threadIdx.x, lane = tid & 31;
    const int wid = tid >> 5;
    const int hl = lane >> 4, r = lane & 15;
    const int bm = blockIdx.y * 128, bn = blockIdx.x * 128;
    const int wm = (wid >> 2) * 64;   // 0 / 64
    const int wn = (wid & 3) * 32;    // 0..96

    f32x8 acc[4][2] = {};

    // prologue: stage K-slice 0 into buffer 0
    if (wid == 0) {
        tdm_load_2d(&sAb[0][0], A + (size_t)bm * K, 64, 128,
                    (unsigned)K * 2, GEMM_PAD);
        tdm_load_2d(&sBb[0][0], B + (size_t)bn * K, 64, 128,
                    (unsigned)K * 2, GEMM_PAD);
    }

    int i = 0;
    for (int k0 = 0; k0 < K; k0 += 32, ++i) {
        const int cur = i & 1;
        __syncthreads();   // all waves done reading buf[cur^1] (prev compute)
        if (wid == 0) {
            if (k0 + 32 < K) {
                tdm_load_2d(&sAb[cur ^ 1][0],
                            A + (size_t)bm * K + k0 + 32, 64, 128,
                            (unsigned)K * 2, GEMM_PAD);
                tdm_load_2d(&sBb[cur ^ 1][0],
                            B + (size_t)bn * K + k0 + 32, 64, 128,
                            (unsigned)K * 2, GEMM_PAD);
                __builtin_amdgcn_s_wait_tensorcnt((short)2); // slice i landed
            } else {
                __builtin_amdgcn_s_wait_tensorcnt((short)0);
            }
        }
        __syncthreads();   // broadcast: buf[cur] is ready
#pragma unroll
        for (int mt = 0; mt < 4; ++mt) {
            bf16x16 af = ldsA(&sAb[cur][(wm + mt * 16 + r) * LDK], hl);
#pragma unroll
            for (int nt = 0; nt < 2; ++nt) {
                bf16x16 bf = ldsB(&sBb[cur][(wn + nt * 16 + r) * LDK], hl);
                acc[mt][nt] = wmma_bf16(af, bf, acc[mt][nt]);
            }
        }
    }
    // store: reg rr -> row rr + 8*hl, lane&15 -> col (cols >= N discarded)
#pragma unroll
    for (int mt = 0; mt < 4; ++mt)
#pragma unroll
        for (int nt = 0; nt < 2; ++nt) {
            int col = bn + wn + nt * 16 + r;
            if (col < N) {
#pragma unroll
                for (int rr = 0; rr < 8; ++rr) {
                    int row = bm + wm + mt * 16 + rr + 8 * hl;
                    C[(size_t)row * N + col] = acc[mt][nt][rr];
                }
            }
        }
}

// ---------------------------------------------------------------------------
// Depthwise causal conv1d (K=4) + bias + SiLU over xBC slice of proj.
// ---------------------------------------------------------------------------
__global__ void k_conv_silu(const float* __restrict__ proj,
                            const float* __restrict__ cw,
                            const float* __restrict__ cb,
                            float* __restrict__ xconv,
                            bf16* __restrict__ Bg, bf16* __restrict__ Cg)
{
    int idx = blockIdx.x * 256 + threadIdx.x;
    if (idx >= SEQ * CONV_DIM) return;
    int ch = idx % CONV_DIM, s = idx / CONV_DIM;
    float a = cb[ch];
#pragma unroll
    for (int k = 0; k < CONVK; ++k) {
        int sp = s - (CONVK - 1) + k;
        if (sp >= 0) a += cw[ch * CONVK + k] * proj[(size_t)sp * PROJ_N + INTER + ch];
    }
    a = silu(a);
    if (ch < INTER)              xconv[(size_t)s * INTER + ch] = a;
    else if (ch < INTER + STATE) Bg[s * STATE + (ch - INTER)] = (bf16)a;
    else                         Cg[s * STATE + (ch - INTER - STATE)] = (bf16)a;
}

// ---------------------------------------------------------------------------
// dt = clip(softplus(dt_raw + bias), 0, 100)
// ---------------------------------------------------------------------------
__global__ void k_dt(const float* __restrict__ proj, const float* __restrict__ dtb,
                     float* __restrict__ dtsp)
{
    int idx = blockIdx.x * 256 + threadIdx.x;
    if (idx >= SEQ * NHEADS) return;
    int h = idx & (NHEADS - 1), s = idx >> 6;
    float v = proj[(size_t)s * PROJ_N + INTER + CONV_DIM + h] + dtb[h];
    float sp = v > 20.f ? v : log1pf(__expf(v));
    dtsp[idx] = fminf(fmaxf(sp, 0.f), 100.f);
}

// per (head, chunk): inclusive cumsum of A*dt over the 256-long chunk
__global__ __launch_bounds__(256) void k_dt_cumsum(
    const float* __restrict__ dtsp, const float* __restrict__ Alog,
    float* __restrict__ Acum)
{
    __shared__ float buf[CHUNK];
    int h = blockIdx.x >> 4, c = blockIdx.x & 15;
    int l = threadIdx.x;
    float A = -__expf(Alog[h]);
    buf[l] = A * dtsp[(c * CHUNK + l) * NHEADS + h];
    __syncthreads();
    for (int off = 1; off < CHUNK; off <<= 1) {
        float add = (l >= off) ? buf[l - off] : 0.f;
        __syncthreads();
        buf[l] += add;
        __syncthreads();
    }
    Acum[(h * NCHUNK + c) * CHUNK + l] = buf[l];
}

// ---------------------------------------------------------------------------
// Chunked SSM scan: one workgroup (8 waves) per head, 16 chunks sequential.
// TDM stages the B/C chunk tiles; all einsums are bf16 WMMA on LDS tiles.
// Static LDS = 281 KB (< 320 KB per WGP).
// ---------------------------------------------------------------------------
__global__ __launch_bounds__(256) void k_scan(
    const bf16* __restrict__ Bg, const bf16* __restrict__ Cg,
    const float* __restrict__ xconv, const float* __restrict__ dtsp,
    const float* __restrict__ Acum, const float* __restrict__ Dg,
    float* __restrict__ yout)
{
    __shared__ __align__(16) bf16  sB[CHUNK * STATE];       // B[s][n]        64 KB
    __shared__ __align__(16) bf16  sC[CHUNK * STATE];       // C[s][n]        64 KB
    __shared__ __align__(16) bf16  sXT[HEADDIM * CHUNK];    // (x*dt)^T[p][s] 32 KB
    __shared__ __align__(16) bf16  sBdT[STATE * CHUNK];     // (B*decay)^T    64 KB
    __shared__ __align__(16) float sState[HEADDIM * STATE]; // state[p][n]    32 KB
    __shared__ __align__(16) bf16  sStateBf[HEADDIM * STATE];              // 16 KB
    __shared__ float sAc[CHUNK];                                           //  1 KB
    __shared__ __align__(16) bf16  sM[8][16 * 32];          // per-wave M      8 KB

    const int h   = blockIdx.x;
    const int tid = threadIdx.x, lane = tid & 31, wid = tid >> 5;
    const int hl  = lane >> 4, r = lane & 15;
    const float Dh = Dg[h];

    for (int i = tid; i < HEADDIM * STATE; i += 256) {
        sState[i] = 0.f; sStateBf[i] = (bf16)0.f;
    }
    __syncthreads();

    for (int c = 0; c < NCHUNK; ++c) {
        // ---- stage chunk operands: B/C via TDM, x^T*dt via lanes ----
        if (wid == 0) {
            tdm_load_2d(&sB[0], Bg + (size_t)(c * CHUNK) * STATE,
                        CHUNK, CHUNK, CHUNK, 0u);   // 256 rows x 256 B
            tdm_load_2d(&sC[0], Cg + (size_t)(c * CHUNK) * STATE,
                        CHUNK, CHUNK, CHUNK, 0u);
        }
        sAc[tid] = Acum[(h * NCHUNK + c) * CHUNK + tid];
        for (int i = tid; i < HEADDIM * CHUNK; i += 256) {
            int p = i & (HEADDIM - 1), sl = i >> 6;
            int s = c * CHUNK + sl;
            float v = xconv[(size_t)s * INTER + h * HEADDIM + p] * dtsp[s * NHEADS + h];
            sXT[p * CHUNK + sl] = (bf16)v;
        }
        if (wid == 0) __builtin_amdgcn_s_wait_tensorcnt((short)0);
        __syncthreads();
        const float AcL = sAc[CHUNK - 1];   // chunk total of A*dt
        // B with state-decay, transposed to [n][s]
        for (int i = tid; i < STATE * CHUNK; i += 256) {
            int n = i & (STATE - 1), sl = i >> 7;
            float dec = __expf(AcL - sAc[sl]);
            sBdT[n * CHUNK + sl] = (bf16)((float)sB[sl * STATE + n] * dec);
        }
        __syncthreads();

        // ---- per-l-tile output: Y = Y_off + Y_diag + D*x ----
        bf16* sMw = sM[wid];
        for (int lt = wid; lt < 16; lt += 8) {
            f32x8 Y[4] = {};
            // Y_off = (C_l . state^T), K = n = 128
#pragma unroll
            for (int kt = 0; kt < 4; ++kt) {
                bf16x16 af = ldsA(&sC[(lt * 16 + r) * STATE + kt * 32], hl);
#pragma unroll
                for (int pt = 0; pt < 4; ++pt) {
                    bf16x16 bf = ldsB(&sStateBf[(pt * 16 + r) * STATE + kt * 32], hl);
                    Y[pt] = wmma_bf16(af, bf, Y[pt]);
                }
            }
#pragma unroll
            for (int rr = 0; rr < 8; ++rr) {      // * exp(A_cum[l])
                float sc = __expf(sAc[lt * 16 + rr + 8 * hl]);
#pragma unroll
                for (int pt = 0; pt < 4; ++pt) Y[pt][rr] *= sc;
            }
            // Y_diag: G via WMMA, mask with L = exp(segsum), reuse as A-operand
            for (int sp = 0; sp < 8; ++sp) {
                int s0 = sp * 32;
#pragma unroll
                for (int st = 0; st < 2; ++st) {
                    f32x8 g = {};
#pragma unroll
                    for (int kt = 0; kt < 4; ++kt) {
                        bf16x16 af = ldsA(&sC[(lt * 16 + r) * STATE + kt * 32], hl);
                        bf16x16 bf = ldsB(&sB[(s0 + st * 16 + r) * STATE + kt * 32], hl);
                        g = wmma_bf16(af, bf, g);
                    }
                    int scol = s0 + st * 16 + r;
#pragma unroll
                    for (int rr = 0; rr < 8; ++rr) {
                        int l = lt * 16 + rr + 8 * hl;
                        float v = (scol <= l) ? g[rr] * __expf(sAc[l] - sAc[scol]) : 0.f;
                        sMw[(rr + 8 * hl) * 32 + st * 16 + r] = (bf16)v;
                    }
                }
                asm volatile("s_wait_dscnt 0" ::: "memory");   // intra-wave LDS RAW
                bf16x16 am = ldsA(&sMw[r * 32], hl);
#pragma unroll
                for (int pt = 0; pt < 4; ++pt) {
                    bf16x16 bf = ldsB(&sXT[(pt * 16 + r) * CHUNK + s0], hl);
                    Y[pt] = wmma_bf16(am, bf, Y[pt]);
                }
            }
            // epilogue: + D*x_conv, store y[s][h*64+p]
#pragma unroll
            for (int pt = 0; pt < 4; ++pt) {
                int p = pt * 16 + r;
#pragma unroll
                for (int rr = 0; rr < 8; ++rr) {
                    int l = lt * 16 + rr + 8 * hl;
                    int s = c * CHUNK + l;
                    float xv = xconv[(size_t)s * INTER + h * HEADDIM + p];
                    yout[(size_t)s * INTER + h * HEADDIM + p] = Y[pt][rr] + Dh * xv;
                }
            }
        }
        __syncthreads();

        // ---- state update: state = exp(T_c)*state + x^T . B_decay ----
        float decC = __expf(AcL);
        for (int t = wid; t < 32; t += 8) {
            int pt = t >> 3, nt = t & 7;
            f32x8 acc = {};
#pragma unroll
            for (int kt = 0; kt < 8; ++kt) {       // K = s = 256
                bf16x16 af = ldsA(&sXT[(pt * 16 + r) * CHUNK + kt * 32], hl);
                bf16x16 bf = ldsB(&sBdT[(nt * 16 + r) * CHUNK + kt * 32], hl);
                acc = wmma_bf16(af, bf, acc);
            }
#pragma unroll
            for (int rr = 0; rr < 8; ++rr) {
                int p = pt * 16 + rr + 8 * hl, n = nt * 16 + r;
                int idx = p * STATE + n;
                float v = decC * sState[idx] + acc[rr];
                sState[idx] = v;
                sStateBf[idx] = (bf16)v;
            }
        }
        __syncthreads();
    }
}

// ---------------------------------------------------------------------------
// y = y * silu(gate); RMS-norm over INTER; scale by norm_w; emit bf16
// ---------------------------------------------------------------------------
__global__ __launch_bounds__(256) void k_gate_norm(
    const float* __restrict__ proj, const float* __restrict__ yin,
    const float* __restrict__ nw, bf16* __restrict__ ybf)
{
    __shared__ float red[256];
    const int s = blockIdx.x, t = threadIdx.x;
    float vals[16];
    float ss = 0.f;
#pragma unroll
    for (int i = 0; i < 16; ++i) {
        int j = t + i * 256;
        float g = silu(proj[(size_t)s * PROJ_N + j]);
        float v = yin[(size_t)s * INTER + j] * g;
        vals[i] = v; ss += v * v;
    }
    red[t] = ss; __syncthreads();
    for (int off = 128; off > 0; off >>= 1) {
        if (t < off) red[t] += red[t + off];
        __syncthreads();
    }
    float inv = rsqrtf(red[0] / (float)INTER + 1e-5f);
#pragma unroll
    for (int i = 0; i < 16; ++i) {
        int j = t + i * 256;
        ybf[(size_t)s * INTER + j] = (bf16)(vals[i] * inv * nw[j]);
    }
}

// ---------------------------------------------------------------------------
extern "C" void kernel_launch(void* const* d_in, const int* in_sizes, int n_in,
                              void* d_out, int out_size, void* d_ws, size_t ws_size,
                              hipStream_t stream)
{
    const float* hs    = (const float*)d_in[0];   // [1,4096,2048]
    const float* w1    = (const float*)d_in[1];   // [8512,2048]
    const float* convw = (const float*)d_in[2];   // [4352,1,4]
    const float* convb = (const float*)d_in[3];   // [4352]
    const float* dtb   = (const float*)d_in[4];   // [64]
    const float* Alog  = (const float*)d_in[5];   // [64]
    const float* Dg    = (const float*)d_in[6];   // [64]
    const float* nw    = (const float*)d_in[7];   // [4096]
    const float* w2    = (const float*)d_in[8];   // [2048,4096]
    float* out = (float*)d_out;                   // [1,4096,2048]

    char* ws = (char*)d_ws;
    size_t off = 0;
    auto carve = [&](size_t bytes) { char* p = ws + off; off += (bytes + 255) & ~(size_t)255; return p; };
    bf16*  hs_bf = (bf16*) carve((size_t)SEQ * HIDDEN * 2);
    bf16*  w1_bf = (bf16*) carve((size_t)PROJ_N * HIDDEN * 2);
    bf16*  w2_bf = (bf16*) carve((size_t)HIDDEN * INTER * 2);
    float* proj  = (float*)carve((size_t)SEQ * PROJ_N * 4);
    float* xconv = (float*)carve((size_t)SEQ * INTER * 4);
    bf16*  Bg    = (bf16*) carve((size_t)SEQ * STATE * 2);
    bf16*  Cg    = (bf16*) carve((size_t)SEQ * STATE * 2);
    float* dtsp  = (float*)carve((size_t)SEQ * NHEADS * 4);
    float* Acum  = (float*)carve((size_t)NHEADS * NCHUNK * CHUNK * 4);
    float* yg    = (float*)carve((size_t)SEQ * INTER * 4);
    bf16*  ybf   = (bf16*) carve((size_t)SEQ * INTER * 2);

    // 1. fp32 -> bf16 operands
    k_to_bf16<<<(SEQ * HIDDEN + 255) / 256, 256, 0, stream>>>(hs, hs_bf, SEQ * HIDDEN);
    k_to_bf16<<<(PROJ_N * HIDDEN + 255) / 256, 256, 0, stream>>>(w1, w1_bf, PROJ_N * HIDDEN);
    k_to_bf16<<<(HIDDEN * INTER + 255) / 256, 256, 0, stream>>>(w2, w2_bf, HIDDEN * INTER);

    // 2. in_proj GEMM: proj[4096][8512]
    k_gemm_bf16<<<dim3((PROJ_N + 127) / 128, SEQ / 128), 256, 0, stream>>>(
        hs_bf, w1_bf, proj, SEQ, PROJ_N, HIDDEN);

    // 3. depthwise causal conv + SiLU
    k_conv_silu<<<(SEQ * CONV_DIM + 255) / 256, 256, 0, stream>>>(
        proj, convw, convb, xconv, Bg, Cg);

    // 4. dt softplus + per-chunk cumsum of A*dt
    k_dt<<<(SEQ * NHEADS + 255) / 256, 256, 0, stream>>>(proj, dtb, dtsp);
    k_dt_cumsum<<<NHEADS * NCHUNK, 256, 0, stream>>>(dtsp, Alog, Acum);

    // 5. chunked SSM scan (one WGP per head)
    k_scan<<<NHEADS, 256, 0, stream>>>(Bg, Cg, xconv, dtsp, Acum, Dg, yg);

    // 6. gating + RMS norm -> bf16
    k_gate_norm<<<SEQ, 256, 0, stream>>>(proj, yg, nw, ybf);

    // 7. out_proj GEMM -> d_out
    k_gemm_bf16<<<dim3(HIDDEN / 128, SEQ / 128), 256, 0, stream>>>(
        ybf, w2_bf, out, SEQ, HIDDEN, INTER);

    (void)in_sizes; (void)n_in; (void)out_size; (void)ws_size;
}
